// EmbeddingGateMemoryModel_17789754540435
// MI455X (gfx1250) — matrix operand order repaired and checked by reading, not compiled, not run on hardware
//
#include <hip/hip_runtime.h>
#include <hip/hip_bf16.h>
#include <stdint.h>

// ---------------- problem constants ----------------
#define BB   128
#define TT   1024
#define INND 64
#define HH   512
#define GG   2560          // 5*H
#define EE   64
#define VV   1000

#define HSTR 520           // padded LDS row stride (halfs) for h  (1040B = 65*16B, 4-bank skew)
#define XSTR 72            // padded LDS row stride (halfs) for x/emb staging

#define WHGB (HH * HH * 2)     // byte stride between gate blocks in whT (524288)
#define WXGB (HH * INND * 2)   // byte stride between gate blocks in wxT (65536)

typedef __attribute__((ext_vector_type(16))) __bf16 v16bf;
typedef __attribute__((ext_vector_type(8)))  float  v8f;

union Frag { uint4 q[2]; v16bf v; };

// fp32 -> bf16 round-to-nearest-even
__device__ __forceinline__ __bf16 f2bf(float f) {
    unsigned u = __builtin_bit_cast(unsigned, f);
    u = u + 0x7FFFu + ((u >> 16) & 1u);
    unsigned short s = (unsigned short)(u >> 16);
    return __builtin_bit_cast(__bf16, s);
}

__device__ __forceinline__ float fsig(float v)  { return 1.0f / (1.0f + __expf(-v)); }
__device__ __forceinline__ float ftanh(float v) {
    float a = fabsf(v);
    float e = __expf(-2.0f * a);
    float r = (1.0f - e) / (1.0f + e);
    return copysignf(r, v);
}

// A operand (16x32 bf16, M = lane&15). rowl already includes the hi*8 lane-half skew;
// k0 folds into ds_load_b128 offset:imm.
__device__ __forceinline__ v16bf load_a2(const __bf16* rowl, int k0) {
    Frag f;
    f.q[0] = *(const uint4*)(rowl + k0);
    f.q[1] = *(const uint4*)(rowl + k0 + 16);
    return f.v;
}
// B operand (32x16 bf16): uniform scalar base + per-lane byte offset + constant byte offset.
__device__ __forceinline__ v16bf load_b_off(const char* base, int voff, int uconst) {
    Frag f;
    f.q[0] = *(const uint4*)(base + voff + uconst);
    f.q[1] = *(const uint4*)(base + voff + uconst + 32);
    return f.v;
}

__device__ __forceinline__ v8f wmma_bf16(v16bf a, v16bf b, v8f c) {
    return __builtin_amdgcn_wmma_f32_16x16x32_bf16(false, a, false, b, (short)0, c, false, false);
}

// ---------------- weight prep: fp32 -> bf16, transposed for contiguous K runs ----------------
// whT[n][k] : [2560][512], wxT[n][k] : [2560][64], weT[n][k] : [512][64], embB[v][k] : [1000][64]
__global__ void prep_weights_kernel(const float* __restrict__ Wx, const float* __restrict__ Wh,
                                    const float* __restrict__ We, const float* __restrict__ emb,
                                    __bf16* __restrict__ wxT, __bf16* __restrict__ whT,
                                    __bf16* __restrict__ weT, __bf16* __restrict__ embB) {
    int i = blockIdx.x * blockDim.x + threadIdx.x;
    if (i < HH * GG) {                       // W_h [512][2560] -> whT [2560][512]
        int k = i / GG, n = i % GG;
        whT[(size_t)n * HH + k] = f2bf(Wh[i]);
        return;
    }
    int j = i - HH * GG;
    if (j < INND * GG) {                     // W_x [64][2560] -> wxT [2560][64]
        int k = j / GG, n = j % GG;
        wxT[(size_t)n * INND + k] = f2bf(Wx[j]);
        return;
    }
    int l = j - INND * GG;
    if (l < EE * HH) {                       // W_e [64][512] -> weT [512][64]
        int k = l / HH, n = l % HH;
        weT[(size_t)n * EE + k] = f2bf(We[l]);
        return;
    }
    int m = l - EE * HH;
    if (m < VV * EE) embB[m] = f2bf(emb[m]); // embedding table, row-major (A operand)
}

// ---------------- persistent fused recurrence ----------------
// grid = 8 (batch tiles of 16), block = 512 (16 waves = 4/SIMD -> ~256-VGPR budget, no spills);
// wave w owns h-column groups {w, w+16}, processed sequentially so acc registers are reused.
__global__ __launch_bounds__(512, 1) void lstm_persistent_kernel(
    const int*   __restrict__ eidx, const float* __restrict__ x,
    const float* __restrict__ bg,   const float* __restrict__ beb,
    const float* __restrict__ wlin, const float* __restrict__ blinp,
    const __bf16* __restrict__ whT, const __bf16* __restrict__ wxT,
    const __bf16* __restrict__ weT, const __bf16* __restrict__ embB,
    float* __restrict__ out)
{
    __shared__ __bf16 h_lds[2][16 * HSTR];   // h state, bf16, double-buffered
    __shared__ __bf16 x_lds[16 * XSTR];      // x[:,t,:] tile (bf16)
    __shared__ __bf16 e_lds[16 * XSTR];      // embedding[e[:,t]] tile (bf16)
    __shared__ float  o_acc[2][16];          // fused h @ W_lin reduction

    const int tid  = threadIdx.x;
    const int w    = tid >> 5;               // wave id 0..15
    const int lane = tid & 31;
    const int N    = lane & 15;              // WMMA N / M index within lane group
    const int hi   = lane >> 4;              // lane half select
    const int r0   = blockIdx.x * 16;        // batch tile base row

    // init: h0 = 0, output accumulators = 0
    for (int i = tid; i < 16 * HSTR; i += 512) h_lds[0][i] = f2bf(0.0f);
    if (tid < 32) o_acc[tid >> 4][tid & 15] = 0.0f;

    const float blin = blinp[0];

    // per-column-group constants (cg = 0 -> group w, cg = 1 -> group w+16)
    float wlv[2], beLv[2], bgv[2][5];
    int   voffW[2], voffX[2], voffE[2];
#pragma unroll
    for (int cg = 0; cg < 2; ++cg) {
        const int col = (w + 16 * cg) * 16 + N;          // this lane's gate column
        wlv[cg]  = wlin[col];
        beLv[cg] = beb[col];
#pragma unroll
        for (int g = 0; g < 5; ++g) bgv[cg][g] = bg[g * HH + col];
        voffW[cg] = (col * HH   + hi * 8) * 2;           // < 2.63 MB, fits i32
        voffX[cg] = (col * INND + hi * 8) * 2;
        voffE[cg] = (col * EE   + hi * 8) * 2;
    }

    v8f c0 = {0.f, 0.f, 0.f, 0.f, 0.f, 0.f, 0.f, 0.f};  // cell state (group 0), persistent
    v8f c1 = {0.f, 0.f, 0.f, 0.f, 0.f, 0.f, 0.f, 0.f};  // cell state (group 1), persistent

    const char* whB = (const char*)whT;
    const char* wxB = (const char*)wxT;
    const char* weB = (const char*)weT;

    __syncthreads();

    for (int t = 0; t < TT; ++t) {
        // ---- stage phase: drain previous output, zero this step's reducer, load x/emb tiles
        if (t > 0 && tid < 16)
            out[(size_t)(r0 + tid) * TT + (t - 1)] = o_acc[(t - 1) & 1][tid] + blin;
        if (tid >= 16 && tid < 32) o_acc[t & 1][tid - 16] = 0.0f;
#pragma unroll
        for (int i = tid; i < 16 * 64; i += 512) {
            int b = i >> 6, k = i & 63;
            x_lds[b * XSTR + k] = f2bf(x[((size_t)(r0 + b) * TT + t) * INND + k]);
            int row = eidx[(size_t)(r0 + b) * TT + t];
            e_lds[b * XSTR + k] = f2bf(embB[(size_t)row * EE + k]);
        }
        __syncthreads();

        const __bf16* hbl = &h_lds[t & 1][N * HSTR + hi * 8];  // A row = lane&15
        __bf16* hn = &h_lds[(t + 1) & 1][0];

        // ---- two column groups processed sequentially: acc registers reused between them
#pragma unroll 1
        for (int cg = 0; cg < 2; ++cg) {
            const int vW = voffW[cg], vX = voffX[cg], vE = voffE[cg];

            v8f acc0 = {bgv[cg][0],bgv[cg][0],bgv[cg][0],bgv[cg][0],bgv[cg][0],bgv[cg][0],bgv[cg][0],bgv[cg][0]};
            v8f acc1 = {bgv[cg][1],bgv[cg][1],bgv[cg][1],bgv[cg][1],bgv[cg][1],bgv[cg][1],bgv[cg][1],bgv[cg][1]};
            v8f acc2 = {bgv[cg][2],bgv[cg][2],bgv[cg][2],bgv[cg][2],bgv[cg][2],bgv[cg][2],bgv[cg][2],bgv[cg][2]};
            v8f acc3 = {bgv[cg][3],bgv[cg][3],bgv[cg][3],bgv[cg][3],bgv[cg][3],bgv[cg][3],bgv[cg][3],bgv[cg][3]};
            v8f acc4 = {bgv[cg][4],bgv[cg][4],bgv[cg][4],bgv[cg][4],bgv[cg][4],bgv[cg][4],bgv[cg][4],bgv[cg][4]};
            v8f ae   = {beLv[cg],beLv[cg],beLv[cg],beLv[cg],beLv[cg],beLv[cg],beLv[cg],beLv[cg]};

            // gates += h @ W_h   (K = 512)
#pragma unroll 4
            for (int kk = 0; kk < 16; ++kk) {
                const int kb = kk * 64;                        // byte offset of this K-step
                v16bf A  = load_a2(hbl, kk * 32);              // 1 LDS A-frag shared by 5 gates
                v16bf B0 = load_b_off(whB, vW, 0 * WHGB + kb); // grouped distinct B-frags ->
                v16bf B1 = load_b_off(whB, vW, 1 * WHGB + kb); //   clause issue + partial waits
                v16bf B2 = load_b_off(whB, vW, 2 * WHGB + kb);
                v16bf B3 = load_b_off(whB, vW, 3 * WHGB + kb);
                v16bf B4 = load_b_off(whB, vW, 4 * WHGB + kb);
                acc0 = wmma_bf16(A, B0, acc0);
                acc1 = wmma_bf16(A, B1, acc1);
                acc2 = wmma_bf16(A, B2, acc2);
                acc3 = wmma_bf16(A, B3, acc3);
                acc4 = wmma_bf16(A, B4, acc4);
            }
            // gates += x @ W_x (K = 64, fused x_proj);  ae += emb @ W_e (fused emb_t)
#pragma unroll
            for (int kk = 0; kk < 2; ++kk) {
                const int kb = kk * 64;
                v16bf A  = load_a2(x_lds + N * XSTR + hi * 8, kk * 32);
                v16bf B0 = load_b_off(wxB, vX, 0 * WXGB + kb);
                v16bf B1 = load_b_off(wxB, vX, 1 * WXGB + kb);
                v16bf B2 = load_b_off(wxB, vX, 2 * WXGB + kb);
                v16bf B3 = load_b_off(wxB, vX, 3 * WXGB + kb);
                v16bf B4 = load_b_off(wxB, vX, 4 * WXGB + kb);
                v16bf Ae = load_a2(e_lds + N * XSTR + hi * 8, kk * 32);
                v16bf Be = load_b_off(weB, vE, kb);
                acc0 = wmma_bf16(A, B0, acc0);
                acc1 = wmma_bf16(A, B1, acc1);
                acc2 = wmma_bf16(A, B2, acc2);
                acc3 = wmma_bf16(A, B3, acc3);
                acc4 = wmma_bf16(A, B4, acc4);
                ae   = wmma_bf16(Ae, Be, ae);
            }

            // ---- lane-local cell update (C/D layout: VGPR k -> M = k + 8*hi)
            const float wl = wlv[cg];
            const int   colb = (w + 16 * cg) * 16 + N;
#pragma unroll
            for (int k = 0; k < 8; ++k) {
                float cprev = (cg == 0) ? c0[k] : c1[k];
                float ig = fsig(acc0[k]);
                float fg = fsig(acc1[k]);
                float gg = ftanh(acc2[k]);
                float og = fsig(acc3[k]);
                float eg = fsig(acc4[k]);
                float et = ftanh(ae[k]);
                float cv = fg * cprev + ig * gg + eg * et;
                if (cg == 0) c0[k] = cv; else c1[k] = cv;
                float hv = og * ftanh(cv);
                hn[(k + 8 * hi) * HSTR + colb] = f2bf(hv);     // next h (bf16) to LDS
                atomicAdd(&o_acc[t & 1][k + 8 * hi], hv * wl); // fused h @ W_lin
            }
        }
        __syncthreads();
    }

    if (tid < 16)
        out[(size_t)(r0 + tid) * TT + (TT - 1)] = o_acc[(TT - 1) & 1][tid] + blin;
}

// ---------------- launch ----------------
extern "C" void kernel_launch(void* const* d_in, const int* in_sizes, int n_in,
                              void* d_out, int out_size, void* d_ws, size_t ws_size,
                              hipStream_t stream) {
    const int*   e    = (const int*)  d_in[0];
    const float* x    = (const float*)d_in[1];
    const float* emb  = (const float*)d_in[2];
    const float* Wx   = (const float*)d_in[3];
    const float* Wh   = (const float*)d_in[4];
    const float* We   = (const float*)d_in[5];
    const float* bg   = (const float*)d_in[6];
    const float* be   = (const float*)d_in[7];
    const float* Wlin = (const float*)d_in[8];
    const float* blin = (const float*)d_in[9];

    char* ws = (char*)d_ws;
    size_t off = 0;
    __bf16* whT  = (__bf16*)(ws + off); off += (size_t)GG * HH * 2;     // 2.62 MB
    __bf16* wxT  = (__bf16*)(ws + off); off += (size_t)GG * INND * 2;   // 320 KB
    __bf16* weT  = (__bf16*)(ws + off); off += (size_t)HH * EE * 2;     // 64 KB
    __bf16* embB = (__bf16*)(ws + off); off += (size_t)VV * EE * 2;     // 125 KB

    const int prep_n = HH * GG + INND * GG + EE * HH + VV * EE;
    prep_weights_kernel<<<(prep_n + 255) / 256, 256, 0, stream>>>(
        Wx, Wh, We, emb, wxT, whT, weT, embB);

    lstm_persistent_kernel<<<BB / 16, 512, 0, stream>>>(
        e, x, bg, be, Wlin, blin, whT, wxT, weT, embB, (float*)d_out);
}
